// Discriminator_78889959293137
// MI455X (gfx1250) — matrix operand order repaired and checked by reading, not compiled, not run on hardware
//
#include <hip/hip_runtime.h>
#include <stdint.h>

#define T_LEN 1024
#define B_SZ  64
#define DH_   256
#define K_    8
#define H_    512
#define G3_   1536
#define MROWS (T_LEN * B_SZ)   // 65536
#define WSTRIDE 514            // padded bf16 row stride in LDS (bank-conflict-free)

typedef __attribute__((ext_vector_type(16))) __bf16 v16bf;
typedef __attribute__((ext_vector_type(8)))  float  v8f;

union ABf { v16bf v; uint32_t u[8]; };
union Cf  { v8f   v; float    f[8]; };

__device__ __forceinline__ uint16_t f2bf(float f) {
  uint32_t u = __builtin_bit_cast(uint32_t, f);
  u += 0x7FFFu + ((u >> 16) & 1u);
  return (uint16_t)(u >> 16);
}

// 16-bit A-matrix 16x32 layout: lane half hi, VGPR j -> K index (pairs), contiguous k = 16*hi + 2*j
__device__ __forceinline__ int a_kidx(int j, int hi) {
  return 2 * j + hi * 16;
}

__device__ __forceinline__ v8f wmma_bf(const ABf& a, const ABf& b, v8f c) {
  return __builtin_amdgcn_wmma_f32_16x16x32_bf16(false, a.v, false, b.v,
                                                 (short)0, c, false, false);
}

// ---------------------------------------------------------------- prep ----
__global__ void k_prep(const float* __restrict__ Wih, const float* __restrict__ Whh,
                       uint16_t* __restrict__ wih_bf, uint16_t* __restrict__ whh_bf,
                       float* __restrict__ racc, uint32_t* __restrict__ hbuf32,
                       uint32_t* __restrict__ bar) {
  int i = blockIdx.x * 256 + threadIdx.x;
  if (i < G3_ * H_) { wih_bf[i] = f2bf(Wih[i]); whh_bf[i] = f2bf(Whh[i]); }
  if (i < MROWS)          racc[i] = 0.0f;
  if (i < (B_SZ * H_ / 2)) hbuf32[i] = 0u;   // h0 = 0
  if (i == 0)             *bar = 0u;
}

// ------------------------------------------------- x = [s | sum_k p] ------
__global__ void k_build_x(const float* __restrict__ s, const float* __restrict__ p,
                          uint16_t* __restrict__ xbf) {
  size_t idx = (size_t)blockIdx.x * 256 + threadIdx.x;  // < MROWS*H_
  int h   = (int)(idx & (H_ - 1));
  int row = (int)(idx >> 9);          // row = t*64 + b
  int t = row >> 6, b = row & 63;
  float v;
  if (h < DH_) {
    v = s[((size_t)b * T_LEN + t) * DH_ + h];
  } else {
    int hc = h - DH_;
    const float* pp = p + (((size_t)b * T_LEN + t) * K_) * DH_ + hc;
    v = 0.f;
#pragma unroll
    for (int k = 0; k < K_; ++k) v += pp[k * DH_];
  }
  xbf[idx] = f2bf(v);
}

// ------------------------------------- gi = x * W_ih^T  (bf16 WMMA) -------
__global__ void __launch_bounds__(256)
k_gi(const uint16_t* __restrict__ xbf, const uint16_t* __restrict__ wih_bf,
     float* __restrict__ gi) {
  int lane = threadIdx.x & 31, wave = threadIdx.x >> 5;
  int wid   = blockIdx.x * 8 + wave;
  int mtile = wid >> 3;               // 4096 M-tiles of 16 rows
  int cbase = (wid & 7) * 192;        // 8 column groups of 12 N-tiles
  int nl = lane & 15, hi = lane >> 4;

  Cf acc[12];
#pragma unroll
  for (int n = 0; n < 12; ++n)
#pragma unroll
    for (int e = 0; e < 8; ++e) acc[n].f[e] = 0.f;

  const uint16_t* arow = xbf + (size_t)(mtile * 16 + nl) * H_;
  for (int kc = 0; kc < 16; ++kc) {
    ABf A;
#pragma unroll
    for (int j = 0; j < 8; ++j)
      A.u[j] = *(const uint32_t*)(arow + kc * 32 + a_kidx(j, hi));
#pragma unroll
    for (int n = 0; n < 12; ++n) {
      ABf Bm;
      const uint16_t* brow =
          wih_bf + (size_t)(cbase + n * 16 + nl) * H_ + kc * 32 + hi * 16;
#pragma unroll
      for (int j = 0; j < 8; ++j) Bm.u[j] = *(const uint32_t*)(brow + 2 * j);
      acc[n].v = wmma_bf(A, Bm, acc[n].v);
    }
  }
#pragma unroll
  for (int n = 0; n < 12; ++n) {
    int col = cbase + n * 16 + nl;
#pragma unroll
    for (int r = 0; r < 8; ++r)
      gi[(size_t)(mtile * 16 + hi * 8 + r) * G3_ + col] = acc[n].f[r];
  }
}

// ----------------------------- persistent GRU recurrence + reward ---------
__global__ void __launch_bounds__(512)
k_recur(const float* __restrict__ gi, const uint16_t* __restrict__ whh_bf,
        const float* __restrict__ bih, const float* __restrict__ bhh,
        const float* __restrict__ Wt, uint16_t* __restrict__ hbuf,
        float* __restrict__ racc, uint32_t* bar) {
  extern __shared__ uint16_t lds[];
  uint16_t* Wlds = lds;                        // 192 x WSTRIDE bf16 (192 KB)
  uint16_t* hlds = lds + 192 * WSTRIDE;        //  64 x WSTRIDE bf16 ( 66 KB)
  const int tid   = threadIdx.x;
  const int jBase = blockIdx.x * 64;           // this WG owns h-cols [jBase, jBase+64)

  // load W_hh rows {j, 512+j, 1024+j} for our j-slice into LDS (once)
  {
    const uint32_t* wsrc = (const uint32_t*)whh_bf;
    uint32_t* wdst = (uint32_t*)Wlds;
    for (int i = tid; i < 192 * 256; i += 512) {
      int lr = i >> 8, c = i & 255;
      int gr = (lr < 64) ? (jBase + lr)
             : (lr < 128) ? (512 + jBase + lr - 64)
                          : (1024 + jBase + lr - 128);
      wdst[lr * 257 + c] = wsrc[gr * 256 + c];
    }
  }

  const int lane = tid & 31, wave = tid >> 5;
  const int nl = lane & 15, hi = lane >> 4;
  const int mt = wave >> 2, jt = wave & 3;     // 4 batch-tiles x 4 j-tiles
  const int batchBase = mt * 16;
  const int jglob = jBase + jt * 16 + nl;

  const float bihr = bih[jglob], bihz = bih[512 + jglob], bihn = bih[1024 + jglob];
  const float bhhr = bhh[jglob], bhhz = bhh[512 + jglob], bhhn = bhh[1024 + jglob];
  const float wtj  = Wt[jglob];

  float hprev[8];
#pragma unroll
  for (int r = 0; r < 8; ++r) hprev[r] = 0.f;

  volatile const uint32_t* hb32 = (volatile const uint32_t*)hbuf;
  uint32_t* hl32 = (uint32_t*)hlds;
  __syncthreads();

  for (int t = 0; t < T_LEN; ++t) {
    // full 64x512 bf16 h : global -> LDS (bank-padded)
    for (int i = tid; i < 64 * 256; i += 512) {
      int row = i >> 8, c = i & 255;
      hl32[row * 257 + c] = hb32[row * 256 + c];
    }
    __syncthreads();

    Cf aR, aZ, aN;
#pragma unroll
    for (int r = 0; r < 8; ++r) { aR.f[r] = 0.f; aZ.f[r] = 0.f; aN.f[r] = 0.f; }

    const uint16_t* harow = hlds + (size_t)(batchBase + nl) * WSTRIDE;
    for (int kc = 0; kc < 16; ++kc) {
      ABf A;
#pragma unroll
      for (int j = 0; j < 8; ++j)
        A.u[j] = *(const uint32_t*)(harow + kc * 32 + a_kidx(j, hi));
      const uint16_t* b0 = Wlds + (size_t)(jt * 16 + nl) * WSTRIDE + kc * 32 + hi * 16;
      ABf Bm;
#pragma unroll
      for (int j = 0; j < 8; ++j) Bm.u[j] = *(const uint32_t*)(b0 + 2 * j);
      aR.v = wmma_bf(A, Bm, aR.v);
      const uint16_t* b1 = b0 + 64 * WSTRIDE;
#pragma unroll
      for (int j = 0; j < 8; ++j) Bm.u[j] = *(const uint32_t*)(b1 + 2 * j);
      aZ.v = wmma_bf(A, Bm, aZ.v);
      const uint16_t* b2 = b0 + 128 * WSTRIDE;
#pragma unroll
      for (int j = 0; j < 8; ++j) Bm.u[j] = *(const uint32_t*)(b2 + 2 * j);
      aN.v = wmma_bf(A, Bm, aN.v);
    }

    float part[8];
#pragma unroll
    for (int r = 0; r < 8; ++r) {
      int bglob = batchBase + hi * 8 + r;
      const float* grow = gi + ((size_t)t * 64 + bglob) * G3_;
      float sr = grow[jglob]        + bihr + aR.f[r] + bhhr;
      float sz = grow[512 + jglob]  + bihz + aZ.f[r] + bhhz;
      float hn = aN.f[r] + bhhn;
      float rr = 1.f / (1.f + __expf(-sr));
      float zz = 1.f / (1.f + __expf(-sz));
      float ax = grow[1024 + jglob] + bihn + rr * hn;
      float e2 = __expf(2.f * ax);
      float nn = (e2 - 1.f) / (e2 + 1.f);
      float hnew = (1.f - zz) * nn + zz * hprev[r];
      hprev[r] = hnew;                                   // carry in registers
      hbuf[(size_t)bglob * H_ + jglob] = f2bf(hnew);     // publish for all WGs
      part[r] = wtj * hnew;
    }
    // reward partial dot: reduce over the 16 lanes of each half
#pragma unroll
    for (int m = 1; m < 16; m <<= 1)
#pragma unroll
      for (int r = 0; r < 8; ++r) part[r] += __shfl_xor(part[r], m, 16);
    if (nl == 0) {
#pragma unroll
      for (int r = 0; r < 8; ++r)
        atomicAdd(&racc[(size_t)(batchBase + hi * 8 + r) * T_LEN + t], part[r]);
    }

    // one grid-wide barrier per step (monotonic counter, 8 resident WGs)
    __threadfence();
    __syncthreads();
    if (tid == 0) {
      atomicAdd((unsigned int*)bar, 1u);
      uint32_t target = 8u * (uint32_t)(t + 1);
      volatile uint32_t* vb = (volatile uint32_t*)bar;
      while (*vb < target) __builtin_amdgcn_s_sleep(1);
    }
    __syncthreads();
  }
}

// --------------------------------------------------------- epilogue -------
__global__ void k_out(const float* __restrict__ racc, const float* __restrict__ bt,
                      float* __restrict__ out) {
  int i = blockIdx.x * 256 + threadIdx.x;
  if (i < MROWS) out[i] = 1.f / (1.f + __expf(-(racc[i] + bt[0])));
}

// ---------------------------------------------------------------------------
extern "C" void kernel_launch(void* const* d_in, const int* in_sizes, int n_in,
                              void* d_out, int out_size, void* d_ws, size_t ws_size,
                              hipStream_t stream) {
  const float* s   = (const float*)d_in[0];
  const float* p   = (const float*)d_in[1];
  const float* Wih = (const float*)d_in[2];
  const float* Whh = (const float*)d_in[3];
  const float* bih = (const float*)d_in[4];
  const float* bhh = (const float*)d_in[5];
  const float* Wt  = (const float*)d_in[6];
  const float* bt  = (const float*)d_in[7];
  float* out = (float*)d_out;

  char* ws = (char*)d_ws;
  uint16_t* xbf   = (uint16_t*)(ws);                 //  64 MB  x bf16 [65536][512]
  uint16_t* wihbf = (uint16_t*)(ws + 67108864);      // 1.5 MB
  uint16_t* whhbf = (uint16_t*)(ws + 68681728);      // 1.5 MB
  float*    gi    = (float*)   (ws + 70254592);      // 384 MB  gi f32 [65536][1536]
  uint16_t* hbuf  = (uint16_t*)(ws + 472907776);     //  64 KB  h bf16 [64][512]
  float*    racc  = (float*)   (ws + 472973312);     // 256 KB  reward accum [B][T]
  uint32_t* bar   = (uint32_t*)(ws + 473235456);     // barrier counter

  k_prep<<<(G3_ * H_ + 255) / 256, 256, 0, stream>>>(Wih, Whh, wihbf, whhbf,
                                                     racc, (uint32_t*)hbuf, bar);
  k_build_x<<<(MROWS * H_) / 256, 256, 0, stream>>>(s, p, xbf);
  k_gi<<<4096, 256, 0, stream>>>(xbf, wihbf, gi);

  size_t ldsBytes = (size_t)(192 + 64) * WSTRIDE * sizeof(uint16_t); // ~257 KB
  (void)hipFuncSetAttribute((const void*)k_recur,
                            hipFuncAttributeMaxDynamicSharedMemorySize,
                            (int)ldsBytes);
  k_recur<<<8, 512, ldsBytes, stream>>>(gi, whhbf, bih, bhh, Wt, hbuf, racc, bar);

  k_out<<<(MROWS + 255) / 256, 256, 0, stream>>>(racc, bt, out);
}
